// AttentionHead_65575560675478
// MI455X (gfx1250) — matrix operand order repaired and checked by reading, not compiled, not run on hardware
//
#include <hip/hip_runtime.h>

// CDNA5 / gfx1250 flash-attention head.
// Matrix math on V_WMMA_F32_16X16X32_F16 (wave32, f32 accumulate).
// K tiles staged to LDS with GLOBAL_LOAD_ASYNC_TO_LDS_B128 (double buffered),
// V tiles staged transposed so B-fragments are contiguous LDS vector loads.

typedef _Float16 v16h __attribute__((ext_vector_type(16)));
typedef _Float16 v8h  __attribute__((ext_vector_type(8)));
typedef float    v8f  __attribute__((ext_vector_type(8)));
typedef float    v4f  __attribute__((ext_vector_type(4)));
typedef int      ahv4i __attribute__((vector_size(16)));   // matches builtin's V4i

#define B_  4
#define T_  2048
#define C_  768
#define H_  64

#define VT_STRIDE 40   // padded Vt row stride (halves): bank-conflict-free, 16B aligned

#if defined(__has_builtin)
#if __has_builtin(__builtin_amdgcn_global_load_async_to_lds_b128)
#define AH_HAVE_ASYNC 1
#endif
#endif
#ifndef AH_HAVE_ASYNC
#define AH_HAVE_ASYNC 0
#endif

#define AS1 __attribute__((address_space(1)))
#define AS3 __attribute__((address_space(3)))

// Copy 16 bytes global -> LDS. Async (ASYNCcnt) when available.
__device__ __forceinline__ void ah_copy_b128(const _Float16* g, _Float16* l) {
#if AH_HAVE_ASYNC
    AS1 ahv4i* gp = (AS1 ahv4i*)(unsigned long long)(const void*)g;
    AS3 ahv4i* lp = (AS3 ahv4i*)(unsigned)(unsigned long long)(void*)l;
    __builtin_amdgcn_global_load_async_to_lds_b128(gp, lp, 0, 0);
#else
    *(v8h*)l = *(const v8h*)g;
#endif
}

__device__ __forceinline__ void ah_wait_async() {
#if AH_HAVE_ASYNC
#if __has_builtin(__builtin_amdgcn_s_wait_asynccnt)
    __builtin_amdgcn_s_wait_asynccnt(0);
#else
    asm volatile("s_wait_asynccnt 0x0" ::: "memory");
#endif
#endif
}

// 16 contiguous halves from LDS as a v16h (two 16B ds loads).
__device__ __forceinline__ v16h ah_ldsv16(const _Float16* p) {
    v8h lo = *(const v8h*)p;
    v8h hi = *(const v8h*)(p + 8);
    v16h r;
#pragma unroll
    for (int i = 0; i < 8; ++i) { r[i] = lo[i]; r[8 + i] = hi[i]; }
    return r;
}

// ---------------------------------------------------------------------------
// Kernel 0: W [C,H] f32 -> Wt [H,C] f16 (transpose+convert) so WMMA
// B-fragments become contiguous 32B per-lane loads.
// ---------------------------------------------------------------------------
__global__ void ah_wconv_kernel(const float* __restrict__ W,
                                _Float16* __restrict__ Wt) {
    int idx = blockIdx.x * blockDim.x + threadIdx.x;
    if (idx < C_ * H_) {
        int h = idx / C_;
        int c = idx - h * C_;
        Wt[idx] = (_Float16)W[c * H_ + h];
    }
}

// ---------------------------------------------------------------------------
// Kernel 1: q/k/v = x @ W{q,k,v}, stored f16 [B*T, H].
// grid.x = B*T/16 row tiles; 4 waves per block, wave w -> cols [16w,16w+16).
// ---------------------------------------------------------------------------
__global__ __launch_bounds__(128) void ah_qkv_kernel(
        const float*    __restrict__ x,
        const _Float16* __restrict__ wtq,
        const _Float16* __restrict__ wtk,
        const _Float16* __restrict__ wtv,
        _Float16* __restrict__ q,
        _Float16* __restrict__ k,
        _Float16* __restrict__ v) {
    const int wave = threadIdx.x >> 5;
    const int lane = threadIdx.x & 31;
    const int g    = lane >> 4;
    const int l16  = lane & 15;

    const int rowtile = blockIdx.x;
    const float* xrow = x + (size_t)(rowtile * 16 + l16) * C_;
    const int col     = wave * 16 + l16;

    const _Float16* wq = wtq + (size_t)col * C_;
    const _Float16* wk = wtk + (size_t)col * C_;
    const _Float16* wv = wtv + (size_t)col * C_;

    v8f accq = {}, acck = {}, accv = {};

    for (int kb = 0; kb < C_; kb += 32) {
        // A fragment (16x32 f16): lane<16 K=0..15; lane>=16 K=16..31.
        v4f f0 = *(const v4f*)(xrow + kb + g * 16);
        v4f f1 = *(const v4f*)(xrow + kb + g * 16 + 4);
        v4f f2 = *(const v4f*)(xrow + kb + g * 16 + 8);
        v4f f3 = *(const v4f*)(xrow + kb + g * 16 + 12);
        v16h a;
#pragma unroll
        for (int i = 0; i < 4; ++i) {
            a[i]      = (_Float16)f0[i];
            a[4 + i]  = (_Float16)f1[i];
            a[8 + i]  = (_Float16)f2[i];
            a[12 + i] = (_Float16)f3[i];
        }
        v16h bq = *(const v16h*)(wq + kb + g * 16);
        v16h bk = *(const v16h*)(wk + kb + g * 16);
        v16h bv = *(const v16h*)(wv + kb + g * 16);

        accq = __builtin_amdgcn_wmma_f32_16x16x32_f16(false, a, false, bq,
                                                      (short)0, accq, false, false);
        acck = __builtin_amdgcn_wmma_f32_16x16x32_f16(false, a, false, bk,
                                                      (short)0, acck, false, false);
        accv = __builtin_amdgcn_wmma_f32_16x16x32_f16(false, a, false, bv,
                                                      (short)0, accv, false, false);
    }

#pragma unroll
    for (int r = 0; r < 8; ++r) {
        size_t o = (size_t)(rowtile * 16 + r + g * 8) * H_ + col;
        q[o] = (_Float16)accq[r];
        k[o] = (_Float16)acck[r];
        v[o] = (_Float16)accv[r];
    }
}

// ---------------------------------------------------------------------------
// Kernel 2: flash attention. grid = (T/64, B), 4 waves/block; wave w owns
// query rows [q0blk+16w, +16). All waves march over the same key tiles:
//  - K tile (32x64 f16) staged flat to LDS via async-to-LDS, double buffered
//  - V tile staged TRANSPOSED (Vt[64][VT_STRIDE]) via VGPRs
//  - scores: 4 WMMAs; online softmax in C-layout regs (16-lane shfl_xor)
//  - P -> LDS -> A-layout; O += P@V: 4 WMMAs with contiguous LDS B-frags
// ---------------------------------------------------------------------------
__global__ __launch_bounds__(128) void ah_attn_kernel(
        const _Float16* __restrict__ qg,
        const _Float16* __restrict__ kg,
        const _Float16* __restrict__ vg,
        float* __restrict__ out) {
    __shared__ _Float16 Kbuf[2][32 * 64];          // 8 KB
    __shared__ _Float16 Vtb[2][64 * VT_STRIDE];    // 10 KB
    __shared__ _Float16 ldsP[4][16 * 32];          // 4 KB (per-wave P tiles)

    const int tid  = threadIdx.x;
    const int wave = tid >> 5;
    const int lane = tid & 31;
    const int g    = lane >> 4;
    const int l16  = lane & 15;
    const int b    = blockIdx.y;
    const int q0blk = blockIdx.x * 64;
    const int q0    = q0blk + wave * 16;

    const _Float16* qb = qg + (size_t)b * T_ * H_;
    const _Float16* kb = kg + (size_t)b * T_ * H_;
    const _Float16* vb = vg + (size_t)b * T_ * H_;

    // staging assignments (whole block, 128 threads)
    const int skey = tid & 31;          // V: key row within tile
    const int sh0  = (tid >> 5) * 16;   // V: first h column

    // ---- Q A-fragments, resident for the whole loop ----------------------
    v16h qa0, qa1;
    {
        const _Float16* r = qb + (size_t)(q0 + l16) * H_;
        v8h c0 = *(const v8h*)(r + g * 16);
        v8h c1 = *(const v8h*)(r + 8 + g * 16);
        v8h c2 = *(const v8h*)(r + 32 + g * 16);
        v8h c3 = *(const v8h*)(r + 40 + g * 16);
#pragma unroll
        for (int i = 0; i < 8; ++i) {
            qa0[i]     = c0[i];
            qa0[8 + i] = c1[i];
            qa1[i]     = c2[i];
            qa1[8 + i] = c3[i];
        }
    }

    v8f o[4] = {v8f{}, v8f{}, v8f{}, v8f{}};
    float m[8], l[8];
#pragma unroll
    for (int r = 0; r < 8; ++r) { m[r] = -__builtin_inff(); l[r] = 0.0f; }

    const float scale   = 0.125f;       // 1/sqrt(H)
    const int   kendw   = q0 + 16;      // this wave's exclusive key bound
    const int   kendblk = q0blk + 64;   // block-wide loop bound (uniform)

    // stage tile at key offset j into buffer bi
    auto stage = [&](int bi, int j) {
        // K: flat 4KB copy, 32B per thread, async when available
        {
            const _Float16* src = kb + (size_t)j * H_ + tid * 16;
            _Float16*       dst = &Kbuf[bi][tid * 16];
            ah_copy_b128(src, dst);
            ah_copy_b128(src + 8, dst + 8);
        }
        // V: transpose through VGPRs -> Vt[h][key]
        {
            const _Float16* vs = vb + (size_t)(j + skey) * H_ + sh0;
            v8h a0 = *(const v8h*)vs;
            v8h a1 = *(const v8h*)(vs + 8);
#pragma unroll
            for (int i = 0; i < 8; ++i) {
                Vtb[bi][(sh0 + i) * VT_STRIDE + skey]     = a0[i];
                Vtb[bi][(sh0 + 8 + i) * VT_STRIDE + skey] = a1[i];
            }
        }
    };

    stage(0, 0);
    ah_wait_async();
    __syncthreads();

    int cur = 0;
    for (int j = 0; j < kendblk; j += 32) {
        const int nxt = cur ^ 1;
        if (j + 32 < kendblk) stage(nxt, j + 32);   // overlap DMA with compute

        if (j < kendw) {                            // wave-uniform: EXEC stays full
            // ---- scores: two 16x16 C tiles, 2 WMMAs each ------------------
            const _Float16* Kt = &Kbuf[cur][0];
            v16h b00 = ah_ldsv16(Kt + l16 * 64 + g * 16);
            v16h b01 = ah_ldsv16(Kt + l16 * 64 + 32 + g * 16);
            v16h b10 = ah_ldsv16(Kt + (16 + l16) * 64 + g * 16);
            v16h b11 = ah_ldsv16(Kt + (16 + l16) * 64 + 32 + g * 16);
            v8f s0 = {}, s1 = {};
            s0 = __builtin_amdgcn_wmma_f32_16x16x32_f16(false, qa0, false, b00,
                                                        (short)0, s0, false, false);
            s0 = __builtin_amdgcn_wmma_f32_16x16x32_f16(false, qa1, false, b01,
                                                        (short)0, s0, false, false);
            s1 = __builtin_amdgcn_wmma_f32_16x16x32_f16(false, qa0, false, b10,
                                                        (short)0, s1, false, false);
            s1 = __builtin_amdgcn_wmma_f32_16x16x32_f16(false, qa1, false, b11,
                                                        (short)0, s1, false, false);

            // ---- causal mask + online softmax ----------------------------
            const int key0 = j + l16;
            const int key1 = j + 16 + l16;
#pragma unroll
            for (int r = 0; r < 8; ++r) {
                const int row = q0 + r + g * 8;
                float sc0 = (key0 <= row) ? s0[r] * scale : -__builtin_inff();
                float sc1 = (key1 <= row) ? s1[r] * scale : -__builtin_inff();

                float t = fmaxf(sc0, sc1);
#pragma unroll
                for (int msk = 1; msk < 16; msk <<= 1)
                    t = fmaxf(t, __shfl_xor(t, msk, 32));

                float mn   = fmaxf(m[r], t);
                float corr = __expf(m[r] - mn);
                float p0   = __expf(sc0 - mn);
                float p1   = __expf(sc1 - mn);

                float rs = p0 + p1;
#pragma unroll
                for (int msk = 1; msk < 16; msk <<= 1)
                    rs += __shfl_xor(rs, msk, 32);

                l[r] = l[r] * corr + rs;
                m[r] = mn;
                o[0][r] *= corr; o[1][r] *= corr; o[2][r] *= corr; o[3][r] *= corr;

                // P tile, C layout -> LDS (wave-synchronous)
                ldsP[wave][(r + g * 8) * 32 + l16]      = (_Float16)p0;
                ldsP[wave][(r + g * 8) * 32 + 16 + l16] = (_Float16)p1;
            }

            // ---- P as A fragment -----------------------------------------
            v16h pa;
            {
                const _Float16* pr = &ldsP[wave][l16 * 32];
#pragma unroll
                for (int i = 0; i < 8; ++i) {
                    pa[i]     = pr[g * 16 + i];
                    pa[8 + i] = pr[g * 16 + 8 + i];
                }
            }

            // ---- O += P @ V : contiguous Vt B-fragments ------------------
            const _Float16* Vt = &Vtb[cur][0];
#pragma unroll
            for (int ht = 0; ht < 4; ++ht) {
                v16h bv = ah_ldsv16(Vt + (ht * 16 + l16) * VT_STRIDE + g * 16);
                o[ht] = __builtin_amdgcn_wmma_f32_16x16x32_f16(false, pa, false, bv,
                                                               (short)0, o[ht], false, false);
            }
        }

        ah_wait_async();     // my async stores into buf[nxt] landed
        __syncthreads();     // tile staged by all; all reads of buf[cur] done
        cur = nxt;
    }

    // ---- epilogue: divide by softmax denominator, store f32 --------------
#pragma unroll
    for (int ht = 0; ht < 4; ++ht) {
#pragma unroll
        for (int r = 0; r < 8; ++r) {
            const int row = q0 + r + g * 8;
            out[((size_t)b * T_ + row) * H_ + ht * 16 + l16] = o[ht][r] / l[r];
        }
    }
}

// ---------------------------------------------------------------------------
extern "C" void kernel_launch(void* const* d_in, const int* in_sizes, int n_in,
                              void* d_out, int out_size, void* d_ws, size_t ws_size,
                              hipStream_t stream) {
    const float* x  = (const float*)d_in[0];
    const float* Wq = (const float*)d_in[1];
    const float* Wk = (const float*)d_in[2];
    const float* Wv = (const float*)d_in[3];

    char* ws = (char*)d_ws;
    const size_t wbytes = (size_t)C_ * H_ * sizeof(_Float16);
    _Float16* wtq = (_Float16*)(ws);
    _Float16* wtk = (_Float16*)(ws + wbytes);
    _Float16* wtv = (_Float16*)(ws + 2 * wbytes);
    const size_t qkvelems = (size_t)B_ * T_ * H_;
    _Float16* qh = (_Float16*)(ws + 3 * wbytes);
    _Float16* kh = qh + qkvelems;
    _Float16* vh = kh + qkvelems;

    const int wn = C_ * H_;
    ah_wconv_kernel<<<(wn + 255) / 256, 256, 0, stream>>>(Wq, wtq);
    ah_wconv_kernel<<<(wn + 255) / 256, 256, 0, stream>>>(Wk, wtk);
    ah_wconv_kernel<<<(wn + 255) / 256, 256, 0, stream>>>(Wv, wtv);

    ah_qkv_kernel<<<(B_ * T_) / 16, 128, 0, stream>>>(x, wtq, wtk, wtv, qh, kh, vh);

    ah_attn_kernel<<<dim3(T_ / 64, B_), 128, 0, stream>>>(qh, kh, vh, (float*)d_out);
}
